// HoHyGCNGCN_28604482191971
// MI455X (gfx1250) — matrix operand order repaired and checked by reading, not compiled, not run on hardware
//
#include <hip/hip_runtime.h>

// ---------------- sizes ----------------
#define NBATCH 512
#define NN     200    // nodes
#define NE     100    // edges
#define NHE    39     // hyperedges
#define NHN    23     // hypernodes
#define DIN    64
#define DEIN   16
#define DOUT   128
#define ED     10     // embed dim
#define TD     55     // time dim
#define CI     119    // DIN + TD
#define CIP    128    // CI padded (N dim of support GEMM, mult of 16)
#define KI     357    // 3*CI
#define KIP    384    // KI padded to mult of 32
#define NNP    208    // NN padded to mult of 16
#define KP     224    // 200 padded to mult of 32 (K of support GEMM)
#define CHUNK  64
#define NCHUNK 8

typedef _Float16 half8v  __attribute__((ext_vector_type(8)));
typedef _Float16 half16v __attribute__((ext_vector_type(16)));
typedef float    float8v __attribute__((ext_vector_type(8)));

#define CDIV(a,b) (((a)+(b)-1)/(b))

// 16 contiguous f16 -> half16v (two 16B vector loads)
__device__ inline half16v ld16(const _Float16* p) {
  half8v l = *(const half8v*)p;
  half8v h = *(const half8v*)(p + 8);
  return __builtin_shufflevector(l, h, 0,1,2,3,4,5,6,7,8,9,10,11,12,13,14,15);
}
// A-operand: 8 halves at p, 8 halves at p+16 (ISA 16-bit A layout)
__device__ inline half16v ldA(const _Float16* p) {
  half8v l = *(const half8v*)p;
  half8v h = *(const half8v*)(p + 16);
  return __builtin_shufflevector(l, h, 0,1,2,3,4,5,6,7,8,9,10,11,12,13,14,15);
}

// ======================================================================
// Batch-independent precompute (VALU)
// ======================================================================

// sup[n][m] = row-softmax(relu(E E^T))
__global__ void k_sup(const float* __restrict__ E, float* __restrict__ sup) {
  __shared__ float red[256];
  __shared__ float row[256];
  const int n = blockIdx.x, t = threadIdx.x;
  float v = -1e30f;
  if (t < NN) {
    float acc = 0.f;
    #pragma unroll
    for (int d = 0; d < ED; ++d) acc += E[n*ED+d] * E[t*ED+d];
    v = acc > 0.f ? acc : 0.f;
    row[t] = v;
  }
  red[t] = v; __syncthreads();
  for (int s = 128; s > 0; s >>= 1) { if (t < s) red[t] = fmaxf(red[t], red[t+s]); __syncthreads(); }
  const float mx = red[0]; __syncthreads();
  const float e = (t < NN) ? __expf(row[t] - mx) : 0.f;
  red[t] = e; __syncthreads();
  for (int s = 128; s > 0; s >>= 1) { if (t < s) red[t] += red[t+s]; __syncthreads(); }
  if (t < NN) sup[n*NN+t] = e / red[0];
}

// sup2 = sup @ sup
__global__ void k_sup2(const float* __restrict__ sup, float* __restrict__ sup2) {
  const int id = blockIdx.x*blockDim.x + threadIdx.x;
  if (id >= NN*NN) return;
  const int n = id / NN, m = id % NN;
  float acc = 0.f;
  for (int t = 0; t < NN; ++t) acc += sup[n*NN+t] * sup[t*NN+m];
  sup2[id] = acc;
}

// pack sup/sup2 into padded f16 [2][NNP][KP] (row-major A operand)
__global__ void k_pack_suph(const float* __restrict__ sup, const float* __restrict__ sup2,
                            _Float16* __restrict__ suph) {
  const int id = blockIdx.x*blockDim.x + threadIdx.x;
  if (id >= 2*NNP*KP) return;
  const int k = id / (NNP*KP), r = id % (NNP*KP);
  const int n = r / KP, m = r % KP;
  float v = 0.f;
  if (n < NN && m < NN) v = (k == 0 ? sup : sup2)[n*NN+m];
  suph[id] = (_Float16)v;
}

// per-node dynamic weights, TRANSPOSED for B operand: wnh_t[n][o][ki]
__global__ void k_wnh(const float* __restrict__ E, const float* __restrict__ wp,
                      _Float16* __restrict__ wnh_t) {
  const int id = blockIdx.x*blockDim.x + threadIdx.x;
  if (id >= NN*DIN*KIP) return;
  const int n = id / (DIN*KIP), r = id % (DIN*KIP);
  const int o = r / KIP, ki = r % KIP;
  float acc = 0.f;
  if (ki < KI) {
    const int k = ki / CI, i = ki % CI;
    #pragma unroll
    for (int d = 0; d < ED; ++d) acc += E[n*ED+d] * wp[((d*3 + k)*CI + i)*DIN + o];
  }
  wnh_t[id] = (_Float16)acc;
}

__global__ void k_bias(const float* __restrict__ E, const float* __restrict__ bp,
                       float* __restrict__ bias) {
  const int id = blockIdx.x*blockDim.x + threadIdx.x;
  if (id >= NN*DOUT) return;
  const int n = id / DOUT, o = id % DOUT;
  float acc = 0.f;
  #pragma unroll
  for (int d = 0; d < ED; ++d) acc += E[n*ED+d] * bp[d*DOUT+o];
  bias[id] = acc;
}

// incidence packed f16 [NNP][128] (row-major A operand)
__global__ void k_pack_inch(const float* __restrict__ inc, _Float16* __restrict__ inch) {
  const int id = blockIdx.x*blockDim.x + threadIdx.x;
  if (id >= NNP*128) return;
  const int n = id / 128, m = id % 128;
  inch[id] = (_Float16)((n < NN && m < NE) ? inc[n*NE+m] : 0.f);
}

// ======================================================================
// Deterministic CSR adjacency build (single thread, stable order)
// ======================================================================
__device__ void buildCSR(const int* key, const int* val, int n, int K, int* off, int* lst) {
  int cur[256];
  for (int k = 0; k <= K; ++k) off[k] = 0;
  for (int i = 0; i < n; ++i) off[key[i]+1] += 1;
  for (int k = 0; k < K; ++k) off[k+1] += off[k];
  for (int k = 0; k < K; ++k) cur[k] = off[k];
  for (int i = 0; i < n; ++i) { const int k = key[i]; lst[cur[k]++] = val[i]; }
}

#define A_OFF_HE 0
#define A_LST_HE (A_OFF_HE + NHE + 1)
#define A_OFF_HN (A_LST_HE + 800)
#define A_LST_HN (A_OFF_HN + NN + 1)
#define A_OFF_E2 (A_LST_HN + 800)
#define A_LST_E2 (A_OFF_E2 + NHN + 1)
#define A_OFF_N2 (A_LST_E2 + 400)
#define A_LST_N2 (A_OFF_N2 + NE + 1)
#define A_TOTAL  (A_LST_N2 + 400)

__global__ void k_build_adj(const int* __restrict__ hed, const int* __restrict__ hnd,
                            int* __restrict__ adj) {
  if (threadIdx.x != 0 || blockIdx.x != 0) return;
  buildCSR(hed + 800, hed,       800, NHE, adj + A_OFF_HE, adj + A_LST_HE);
  buildCSR(hed,       hed + 800, 800, NN,  adj + A_OFF_HN, adj + A_LST_HN);
  buildCSR(hnd + 400, hnd,       400, NHN, adj + A_OFF_E2, adj + A_LST_E2);
  buildCSR(hnd,       hnd + 400, 400, NE,  adj + A_OFF_N2, adj + A_LST_N2);
}

// ======================================================================
// Hypergraph mean gathers (deterministic)
// ======================================================================
__global__ void k_gather_mean(const float* __restrict__ src, float* __restrict__ dst,
                              const int* __restrict__ off, const int* __restrict__ lst,
                              int ntgt, int nsrcrows, int nch) {
  const int id = blockIdx.x*blockDim.x + threadIdx.x;
  if (id >= NBATCH*ntgt*nch) return;
  const int b = id / (ntgt*nch), r = id % (ntgt*nch);
  const int t = r / nch, c = r % nch;
  const int s = off[t], e = off[t+1];
  float acc = 0.f;
  for (int p = s; p < e; ++p) acc += src[((size_t)b*nsrcrows + lst[p])*nch + c];
  dst[((size_t)b*ntgt + t)*nch + c] = acc / fmaxf((float)(e - s), 1.f);
}

// channels 64..95: [x @ W_self , h_n @ W_agg] + bias
__global__ void k_nodehyper(const float* __restrict__ x, const float* __restrict__ hn,
                            const float* __restrict__ Wself, const float* __restrict__ Wagg,
                            const float* __restrict__ bias, float* __restrict__ out) {
  const int id = blockIdx.x*blockDim.x + threadIdx.x;
  if (id >= NBATCH*NN*32) return;
  const int b = id / (NN*32), r = id % (NN*32);
  const int n = r / 32, o = r % 32;
  float acc = 0.f;
  if (o < 16) {
    for (int c = 0; c < DIN; ++c) acc += x[((size_t)b*NN + n)*DIN + c] * Wself[c*16 + o];
  } else {
    for (int c = 0; c < DIN; ++c) acc += hn[((size_t)b*NN + n)*DIN + c] * Wagg[c*16 + (o-16)];
  }
  out[((size_t)b*NN + n)*DOUT + 64 + o] = acc + bias[n*DOUT + 64 + o];
}

// x_edge_hyper (f32, 16 ch)
__global__ void k_edgehyper(const float* __restrict__ xe, const float* __restrict__ hn2,
                            const float* __restrict__ Wself, const float* __restrict__ Wagg,
                            float* __restrict__ xeh) {
  const int id = blockIdx.x*blockDim.x + threadIdx.x;
  if (id >= NBATCH*NE*16) return;
  const int b = id / (NE*16), r = id % (NE*16);
  const int e = r / 16, c = r % 16;
  float acc = 0.f;
  if (c < 8) { for (int t = 0; t < DEIN; ++t) acc += xe[((size_t)b*NE + e)*DEIN + t] * Wself[t*8 + c]; }
  else       { for (int t = 0; t < DEIN; ++t) acc += hn2[((size_t)b*NE + e)*DEIN + t] * Wagg[t*8 + (c-8)]; }
  xeh[id] = acc;
}

// tmp = hodge @ x_edge_hyper
__global__ void k_hodgemm(const float* __restrict__ hodge, const float* __restrict__ xeh,
                          float* __restrict__ tmp) {
  const int id = blockIdx.x*blockDim.x + threadIdx.x;
  if (id >= NBATCH*NE*16) return;
  const int b = id / (NE*16), r = id % (NE*16);
  const int m = r / 16, c = r % 16;
  float acc = 0.f;
  for (int t = 0; t < NE; ++t) acc += hodge[m*NE + t] * xeh[((size_t)b*NE + t)*16 + c];
  tmp[id] = acc;
}

// x_e_hh TRANSPOSED f16 for B operand: xehh_t[b][c(32)][m(128)], zeros m>=100
__global__ void k_hh(const float* __restrict__ xeh, const float* __restrict__ tmp,
                     const float* __restrict__ Whoc, const float* __restrict__ bhoc,
                     _Float16* __restrict__ xehh_t) {
  const int id = blockIdx.x*blockDim.x + threadIdx.x;
  if (id >= NBATCH*32*128) return;
  const int b = id / (32*128), r = id % (32*128);
  const int c = r / 128, m = r % 128;
  float v = 0.f;
  if (m < NE) {
    if (c < 16) v = xeh[((size_t)b*NE + m)*16 + c];
    else {
      float acc = bhoc[c-16];
      #pragma unroll
      for (int t = 0; t < 16; ++t) acc += tmp[((size_t)b*NE + m)*16 + t] * Whoc[t*16 + (c-16)];
      v = acc;
    }
  }
  xehh_t[id] = (_Float16)v;
}

// merge_x chunk, TRANSPOSED for B operand: merge_t[bl][i(CIP)][m(KP)]
__global__ void k_merget(const float* __restrict__ x, const float* __restrict__ xt,
                         _Float16* __restrict__ merge_t, int b0) {
  const int id = blockIdx.x*blockDim.x + threadIdx.x;
  if (id >= CHUNK*CIP*KP) return;
  const int bl = id / (CIP*KP), r = id % (CIP*KP);
  const int i = r / KP, m = r % KP;
  float v = 0.f;
  if (m < NN && i < CI)
    v = (i < DIN) ? x[(((size_t)(b0+bl))*NN + m)*DIN + i] : xt[(size_t)(b0+bl)*TD + (i - DIN)];
  merge_t[id] = (_Float16)v;
}

// ybuf[bl][n][KIP]: k=0 segment (ki<119) from x/x_time, tail (ki>=357) zero.
// middle segments are written by k_supgemm afterwards.
__global__ void k_ypack(const float* __restrict__ x, const float* __restrict__ xt,
                        _Float16* __restrict__ ybuf, int b0) {
  const int id = blockIdx.x*blockDim.x + threadIdx.x;
  if (id >= CHUNK*NN*KIP) return;
  const int bl = id / (NN*KIP), r = id % (NN*KIP);
  const int n = r / KIP, ki = r % KIP;
  if (ki < CI) {
    const float v = (ki < DIN) ? x[(((size_t)(b0+bl))*NN + n)*DIN + ki]
                               : xt[(size_t)(b0+bl)*TD + (ki - DIN)];
    ybuf[id] = (_Float16)v;
  } else if (ki >= KI) {
    ybuf[id] = (_Float16)0.0f;
  }
}

// ======================================================================
// WMMA kernels (wave32, multi-tile accumulators, vectorized A/B loads)
// ======================================================================

// sup_k @ merge_x[bl] -> ybuf segment (k+1): 16x64 output per wave (4 N-tiles)
__global__ void __launch_bounds__(32)
k_supgemm(const _Float16* __restrict__ suph, const _Float16* __restrict__ merge_t,
          _Float16* __restrict__ ybuf) {
  const int lane = threadIdx.x, lo = lane & 15, hi = lane >> 4;
  const int nt = blockIdx.x >> 1, itg = blockIdx.x & 1;   // nt 0..12, itg 0..1
  const int k = blockIdx.y, bl = blockIdx.z;
  const _Float16* A = suph + ((size_t)k*NNP + nt*16 + lo)*KP + hi*16;
  const _Float16* Bp[4];
  #pragma unroll
  for (int t = 0; t < 4; ++t)
    Bp[t] = merge_t + (size_t)bl*CIP*KP + (size_t)((itg*4 + t)*16 + lo)*KP + hi*16;
  float8v acc[4];
  #pragma unroll
  for (int t = 0; t < 4; ++t)
    acc[t] = (float8v){0.f,0.f,0.f,0.f,0.f,0.f,0.f,0.f};
  #pragma unroll
  for (int kk = 0; kk < KP; kk += 32) {
    __builtin_prefetch(A + kk + 32, 0, 1);
    const half16v a = ld16(A + kk);
    #pragma unroll
    for (int t = 0; t < 4; ++t) {
      const half16v b = ld16(Bp[t] + kk);
      acc[t] = __builtin_amdgcn_wmma_f32_16x16x32_f16(false, a, false, b, (short)0, acc[t], false, false);
    }
  }
  #pragma unroll
  for (int t = 0; t < 4; ++t) {
    const int i = (itg*4 + t)*16 + lo;
    #pragma unroll
    for (int r = 0; r < 8; ++r) {
      const int n = nt*16 + r + 8*hi;
      if (n < NN && i < CI)
        ybuf[((size_t)bl*NN + n)*KIP + CI + k*CI + i] = (_Float16)acc[t][r];
    }
  }
}

// out[:, n, 0:64] = Y_n(16x384) @ W_n(384x64) + bias, 16x32 per wave (2 N-tiles)
__global__ void __launch_bounds__(32)
k_gconv(const _Float16* __restrict__ ybuf, const _Float16* __restrict__ wnh_t,
        const float* __restrict__ bias, float* __restrict__ out, int b0) {
  const int lane = threadIdx.x, lo = lane & 15, hi = lane >> 4;
  const int bt = blockIdx.x >> 1, og = blockIdx.x & 1;    // bt 0..3, og 0..1
  const int n = blockIdx.y;
  const _Float16* A = ybuf + ((size_t)(bt*16 + lo)*NN + n)*KIP + hi*16;
  const _Float16* Bp[2];
  #pragma unroll
  for (int t = 0; t < 2; ++t)
    Bp[t] = wnh_t + ((size_t)n*DIN + (og*2 + t)*16 + lo)*KIP + hi*16;
  float8v acc[2];
  #pragma unroll
  for (int t = 0; t < 2; ++t)
    acc[t] = (float8v){0.f,0.f,0.f,0.f,0.f,0.f,0.f,0.f};
  #pragma unroll
  for (int kk = 0; kk < KIP; kk += 32) {
    __builtin_prefetch(A + kk + 32, 0, 1);
    const half16v a = ld16(A + kk);
    #pragma unroll
    for (int t = 0; t < 2; ++t) {
      const half16v b = ld16(Bp[t] + kk);
      acc[t] = __builtin_amdgcn_wmma_f32_16x16x32_f16(false, a, false, b, (short)0, acc[t], false, false);
    }
  }
  #pragma unroll
  for (int t = 0; t < 2; ++t) {
    const int ch = (og*2 + t)*16 + lo;
    #pragma unroll
    for (int r = 0; r < 8; ++r) {
      const int bb = b0 + bt*16 + r + 8*hi;
      out[((size_t)bb*NN + n)*DOUT + ch] = acc[t][r] + bias[n*DOUT + ch];
    }
  }
}

// out[:, n, 96:128] = incidence(200x100) @ x_e_hh(100x32) + bias, 16x32 per wave
__global__ void __launch_bounds__(32)
k_incgemm(const _Float16* __restrict__ inch, const _Float16* __restrict__ xehh_t,
          const float* __restrict__ bias, float* __restrict__ out) {
  const int lane = threadIdx.x, lo = lane & 15, hi = lane >> 4;
  const int nt = blockIdx.x;                               // 0..12
  const int b = blockIdx.y;
  const _Float16* A = inch + (size_t)(nt*16 + lo)*128 + hi*16;
  const _Float16* Bp[2];
  #pragma unroll
  for (int t = 0; t < 2; ++t)
    Bp[t] = xehh_t + (size_t)b*32*128 + (size_t)(t*16 + lo)*128 + hi*16;
  float8v acc[2];
  #pragma unroll
  for (int t = 0; t < 2; ++t)
    acc[t] = (float8v){0.f,0.f,0.f,0.f,0.f,0.f,0.f,0.f};
  #pragma unroll
  for (int kk = 0; kk < 128; kk += 32) {
    const half16v a = ld16(A + kk);
    #pragma unroll
    for (int t = 0; t < 2; ++t) {
      const half16v b2 = ld16(Bp[t] + kk);
      acc[t] = __builtin_amdgcn_wmma_f32_16x16x32_f16(false, a, false, b2, (short)0, acc[t], false, false);
    }
  }
  #pragma unroll
  for (int t = 0; t < 2; ++t) {
    #pragma unroll
    for (int r = 0; r < 8; ++r) {
      const int n = nt*16 + r + 8*hi;
      if (n < NN) {
        const int ch = 96 + t*16 + lo;
        out[((size_t)b*NN + n)*DOUT + ch] = acc[t][r] + bias[n*DOUT + ch];
      }
    }
  }
}

// ======================================================================
// host launcher
// ======================================================================
static inline size_t alignup(size_t x) { return (x + 255) & ~(size_t)255; }

extern "C" void kernel_launch(void* const* d_in, const int* in_sizes, int n_in,
                              void* d_out, int out_size, void* d_ws, size_t ws_size,
                              hipStream_t stream) {
  (void)in_sizes; (void)n_in; (void)out_size;
  const float* x     = (const float*)d_in[0];
  const float* xe    = (const float*)d_in[1];
  const float* xt    = (const float*)d_in[2];
  // d_in[3] x_window unused by the reference
  const float* E     = (const float*)d_in[4];
  const float* hodge = (const float*)d_in[5];
  const float* inc   = (const float*)d_in[6];
  const int*   hed   = (const int*)d_in[7];
  const int*   hnd   = (const int*)d_in[8];
  const float* wp    = (const float*)d_in[9];
  const float* bp    = (const float*)d_in[10];
  const float* Wns   = (const float*)d_in[11];
  const float* Wna   = (const float*)d_in[12];
  const float* Wes   = (const float*)d_in[13];
  const float* Wea   = (const float*)d_in[14];
  const float* Whoc  = (const float*)d_in[15];
  const float* bhoc  = (const float*)d_in[16];
  float* out = (float*)d_out;
  char* ws = (char*)d_ws;

  size_t cur = 0;
  auto take = [&](size_t bytes) { size_t o = cur; cur = alignup(cur + bytes); return o; };
  const size_t o_sup    = take((size_t)NN*NN*4);
  const size_t o_sup2   = take((size_t)NN*NN*4);
  const size_t o_suph   = take((size_t)2*NNP*KP*2);
  const size_t o_wnh    = take((size_t)NN*DIN*KIP*2);
  const size_t o_bias   = take((size_t)NN*DOUT*4);
  const size_t o_inch   = take((size_t)NNP*128*2);
  const size_t o_merget = take((size_t)CHUNK*CIP*KP*2);
  const size_t o_ybuf   = take((size_t)CHUNK*NN*KIP*2);
  const size_t o_he     = take((size_t)NBATCH*NHE*DIN*4);
  const size_t o_hn     = take((size_t)NBATCH*NN*DIN*4);
  const size_t o_he2    = take((size_t)NBATCH*NHN*DEIN*4);
  const size_t o_hn2    = take((size_t)NBATCH*NE*DEIN*4);
  const size_t o_xeh    = take((size_t)NBATCH*NE*16*4);
  const size_t o_tmp    = take((size_t)NBATCH*NE*16*4);
  const size_t o_xehh   = take((size_t)NBATCH*32*128*2);
  const size_t o_adj    = take((size_t)A_TOTAL*4);
  if (ws_size < cur) return;

  float*    sup     = (float*)(ws + o_sup);
  float*    sup2    = (float*)(ws + o_sup2);
  _Float16* suph    = (_Float16*)(ws + o_suph);
  _Float16* wnh_t   = (_Float16*)(ws + o_wnh);
  float*    bias    = (float*)(ws + o_bias);
  _Float16* inch    = (_Float16*)(ws + o_inch);
  _Float16* merge_t = (_Float16*)(ws + o_merget);
  _Float16* ybuf    = (_Float16*)(ws + o_ybuf);
  float*    he      = (float*)(ws + o_he);
  float*    hn      = (float*)(ws + o_hn);
  float*    he2     = (float*)(ws + o_he2);
  float*    hn2     = (float*)(ws + o_hn2);
  float*    xeh     = (float*)(ws + o_xeh);
  float*    tmpb    = (float*)(ws + o_tmp);
  _Float16* xehh_t  = (_Float16*)(ws + o_xehh);
  int*      adj     = (int*)(ws + o_adj);

  // ---- batch-independent precompute ----
  k_sup      <<<NN, 256, 0, stream>>>(E, sup);
  k_sup2     <<<CDIV(NN*NN,256), 256, 0, stream>>>(sup, sup2);
  k_pack_suph<<<CDIV(2*NNP*KP,256), 256, 0, stream>>>(sup, sup2, suph);
  k_wnh      <<<CDIV(NN*DIN*KIP,256), 256, 0, stream>>>(E, wp, wnh_t);
  k_bias     <<<CDIV(NN*DOUT,256), 256, 0, stream>>>(E, bp, bias);
  k_pack_inch<<<CDIV(NNP*128,256), 256, 0, stream>>>(inc, inch);
  k_build_adj<<<1, 32, 0, stream>>>(hed, hnd, adj);

  // ---- hypergraph paths (deterministic gathers) ----
  k_gather_mean<<<CDIV(NBATCH*NHE*DIN,256),256,0,stream>>>(x,  he,  adj+A_OFF_HE, adj+A_LST_HE, NHE, NN,  DIN);
  k_gather_mean<<<CDIV(NBATCH*NN*DIN,256), 256,0,stream>>>(he, hn,  adj+A_OFF_HN, adj+A_LST_HN, NN,  NHE, DIN);
  k_gather_mean<<<CDIV(NBATCH*NHN*DEIN,256),256,0,stream>>>(xe, he2, adj+A_OFF_E2, adj+A_LST_E2, NHN, NE,  DEIN);
  k_gather_mean<<<CDIV(NBATCH*NE*DEIN,256), 256,0,stream>>>(he2,hn2, adj+A_OFF_N2, adj+A_LST_N2, NE,  NHN, DEIN);

  k_nodehyper<<<CDIV(NBATCH*NN*32,256), 256, 0, stream>>>(x, hn, Wns, Wna, bias, out);

  k_edgehyper<<<CDIV(NBATCH*NE*16,256), 256, 0, stream>>>(xe, hn2, Wes, Wea, xeh);
  k_hodgemm  <<<CDIV(NBATCH*NE*16,256), 256, 0, stream>>>(hodge, xeh, tmpb);
  k_hh       <<<CDIV(NBATCH*32*128,256), 256, 0, stream>>>(xeh, tmpb, Whoc, bhoc, xehh_t);
  k_incgemm  <<<dim3(13, NBATCH), 32, 0, stream>>>(inch, xehh_t, bias, out);

  // ---- main adaptive-graph conv: chunked WMMA pipeline ----
  for (int c = 0; c < NCHUNK; ++c) {
    const int b0 = c * CHUNK;
    k_merget <<<CDIV(CHUNK*CIP*KP,256), 256, 0, stream>>>(x, xt, merge_t, b0);
    k_ypack  <<<CDIV(CHUNK*NN*KIP,256), 256, 0, stream>>>(x, xt, ybuf, b0);
    k_supgemm<<<dim3(13*2, 2, CHUNK), 32, 0, stream>>>(suph, merge_t, ybuf);
    k_gconv  <<<dim3(4*2, NN), 32, 0, stream>>>(ybuf, wnh_t, bias, out, b0);
  }
}
